// Decoder_44255343018754
// MI455X (gfx1250) — matrix operand order repaired and checked by reading, not compiled, not run on hardware
//
#include <hip/hip_runtime.h>
#include <hip/hip_bf16.h>
#include <stdint.h>

// ---------- types ----------
typedef __attribute__((ext_vector_type(16))) __bf16    v16bf;
typedef __attribute__((ext_vector_type(8)))  uint16_t  v8u;
typedef __attribute__((ext_vector_type(8)))  float     v8f;
typedef __attribute__((ext_vector_type(4)))  int       v4i;
typedef __attribute__((ext_vector_type(8)))  int       v8i;
typedef __attribute__((ext_vector_type(4)))  unsigned int v4u;

struct U8x2 { v8u lo, hi; };
__device__ inline v16bf make_frag(v8u lo, v8u hi) {
  U8x2 t{lo, hi};
  return __builtin_bit_cast(v16bf, t);
}

// ---------- gfx1250 data movers ----------
#if __has_builtin(__builtin_amdgcn_global_load_async_to_lds_b128)
#define USE_ASYNC 1
#else
#define USE_ASYNC 0
#endif

#if __has_builtin(__builtin_amdgcn_tensor_load_to_lds)
#define USE_TDM 1
#else
#define USE_TDM 0
#endif

#define GAS __attribute__((address_space(1)))
#define LAS __attribute__((address_space(3)))

__device__ inline void async_b128(const uint16_t* g, uint16_t* l) {
#if USE_ASYNC
  __builtin_amdgcn_global_load_async_to_lds_b128(
      (GAS v4i*)(uintptr_t)g, (LAS v4i*)(uintptr_t)l, 0, 0);
#else
  *(v8u*)l = *(const v8u*)g;
#endif
}

__device__ inline void wait_async_done() {
#if USE_ASYNC
#if __has_builtin(__builtin_amdgcn_s_wait_asynccnt)
  __builtin_amdgcn_s_wait_asynccnt(0);
#else
  asm volatile("s_wait_asynccnt 0" ::: "memory");
#endif
#endif
}

__device__ inline void wait_tdm_done() {
#if __has_builtin(__builtin_amdgcn_s_wait_tensorcnt)
  __builtin_amdgcn_s_wait_tensorcnt(0);
#else
  asm volatile("s_wait_tensorcnt 0" ::: "memory");
#endif
}

#if USE_TDM
// 2D TDM load: tile = 16 dwords/row x 128 rows, LDS padded 4 dwords per 16 dwords
// (row pitch 80B == PK*2). tensor dims give hardware zero-fill at M/N/K edges.
__device__ inline void tdm_load_2d(uint32_t lds_off, const uint16_t* gaddr,
                                   uint32_t tdim0_dw, uint32_t tdim1_rows,
                                   uint32_t stride_dw)
{
  uint64_t ga = (uint64_t)(uintptr_t)gaddr;
  v4u g0;
  g0[0] = 1u;                                   // count=1, user mode
  g0[1] = lds_off;                              // lds_addr (bytes)
  g0[2] = (uint32_t)ga;                         // global_addr[31:0]
  g0[3] = (uint32_t)(ga >> 32) | (2u << 30);    // global_addr[56:32] | type=2
  v8i g1;
  g1[0] = (int)((2u << 16) | (1u << 20) | (3u << 22) | (3u << 25));
          // data_size=4B, pad_enable, pad_interval=16dw, pad_amount=4dw
  g1[1] = (int)((tdim0_dw & 0xFFFFu) << 16);                    // tensor_dim0 lo
  g1[2] = (int)((tdim0_dw >> 16) | ((tdim1_rows & 0xFFFFu) << 16)); // td0 hi | td1 lo
  g1[3] = (int)(((tdim1_rows >> 16) & 0xFFFFu) | (16u << 16));  // td1 hi | tile_dim0=16
  g1[4] = 128;                                                  // tile_dim1=128, tile_dim2=0
  g1[5] = (int)stride_dw;                                       // tensor_dim0_stride lo
  g1[6] = 0;                                                    // stride hi | dim1_stride lo
  g1[7] = 0;
  v4i z4 = {0, 0, 0, 0};
#if __clang_major__ >= 23
  v8i z8 = {0, 0, 0, 0, 0, 0, 0, 0};
  __builtin_amdgcn_tensor_load_to_lds(g0, g1, z4, z4, z8, 0);
#else
  __builtin_amdgcn_tensor_load_to_lds(g0, g1, z4, z4, 0);
#endif
}
#endif

// ---------- helpers ----------
__device__ inline uint16_t f32_to_bf16_bits(float f) {
  union { float f; uint32_t u; } c; c.f = f;
  uint32_t u = c.u + 0x7FFFu + ((c.u >> 16) & 1u);  // round-to-nearest-even
  return (uint16_t)(u >> 16);
}

// ---------- fused fp32 -> bf16 transpose: out[n][k] = bf16(in[k][n]) ----------
__global__ __launch_bounds__(256) void k_cvt_transpose(
    const float* __restrict__ in, uint16_t* __restrict__ out,
    int K, int N, int ldi, int ldo)
{
  __shared__ float tile[32][33];
  in  += (long long)blockIdx.z * K * N;
  out += (long long)blockIdx.z * K * N;
  int kb = blockIdx.y * 32, nb = blockIdx.x * 32;
  int tx = threadIdx.x & 31, ty = threadIdx.x >> 5;   // 32x8
  for (int r = ty; r < 32; r += 8) {
    int k = kb + r, n = nb + tx;
    tile[r][tx] = (k < K && n < N) ? in[(long long)k * ldi + n] : 0.0f;
  }
  __syncthreads();
  for (int r = ty; r < 32; r += 8) {
    int n = nb + r, k = kb + tx;
    if (n < N && k < K) out[(long long)n * ldo + k] = f32_to_bf16_bits(tile[tx][r]);
  }
}

// ---------- bf16 transpose of V per (b,h): vT[z][d][s] = v[b,s,h*512+d] ----------
__global__ __launch_bounds__(256) void k_transpose_v(
    const uint16_t* __restrict__ in, uint16_t* __restrict__ out)
{
  __shared__ uint16_t tile[32][33];
  int z = blockIdx.z;               // 0..511
  int b = z >> 3, hh = z & 7;
  int sb = blockIdx.y * 32;
  int db = blockIdx.x * 32;
  int tx = threadIdx.x & 31, ty = threadIdx.x >> 5;
  const uint16_t* src = in + ((long long)b * 168) * 4096 + hh * 512;
  for (int r = ty; r < 32; r += 8) {
    int s = sb + r;
    tile[r][tx] = (s < 168) ? src[(long long)s * 4096 + db + tx] : (uint16_t)0;
  }
  __syncthreads();
  uint16_t* dst = out + (long long)z * 512 * 168;
  for (int r = ty; r < 32; r += 8) {
    int d = db + r, s = sb + tx;
    if (s < 168) dst[(long long)d * 168 + s] = tile[tx][r];
  }
}

// ---------- tiled WMMA GEMM (double-buffered LDS, TDM/async staging) ----------
// C[z] = op( (A[z] * B[z]) * scale + bias )
// A: bf16 [M x K] row-major (lda); B: bf16 [N x K] row-major (ldb)
template <bool OUT_BF16, bool RELU>
__global__ __launch_bounds__(256) void k_gemm(
    const uint16_t* __restrict__ A, const uint16_t* __restrict__ B,
    const float* __restrict__ bias, void* __restrict__ Cv,
    int M, int N, int K, int lda, int ldb, int ldc,
    long long sA1, long long sA2, long long sB1, long long sB2,
    long long sC1, long long sC2, float scale)
{
  constexpr int BM = 128, BN = 128, BK = 32;
  constexpr int PK = BK + 8;                 // 80B padded rows, 16B aligned
  __shared__ uint16_t smem[(2 * BM + 2 * BN) * PK];
  uint16_t* sA = smem;                       // two A buffers
  uint16_t* sB = smem + 2 * BM * PK;         // two B buffers, stored [n][k]

  const int z  = blockIdx.z;
  const int z1 = z >> 3, z2 = z & 7;
  const uint16_t* Ab = A + z1 * sA1 + z2 * sA2;
  const uint16_t* Bb = B + z1 * sB1 + z2 * sB2;
  const long long cbase = z1 * sC1 + z2 * sC2;

  const int m0 = blockIdx.y * BM;
  const int n0 = blockIdx.x * BN;
  const int tid  = threadIdx.x;
  const int lane = tid & 31;
  // wave id is wave-uniform: pin it to an SGPR so TDM issue gating is pure SALU
  const int wave = __builtin_amdgcn_readfirstlane(tid >> 5);
  const int wm = wave >> 1;     // 0..3 : 32 rows
  const int wn = wave & 1;      // 0..1 : 64 cols
  const int l15 = lane & 15;
  const int hi  = lane >> 4;

  v8f acc[2][4];
  v8f zacc;
  #pragma unroll
  for (int e = 0; e < 8; ++e) zacc[e] = 0.0f;
  #pragma unroll
  for (int i = 0; i < 2; ++i)
    #pragma unroll
    for (int j = 0; j < 4; ++j) acc[i][j] = zacc;

  const bool fullM = (m0 + BM <= M);
  const bool fullN = (n0 + BN <= N);
  const int ksteps = (K + BK - 1) / BK;

#if USE_TDM
  // one TDM descriptor per tile; hardware zero-fills past tensor dims
  auto stage_pair = [&](int k0, int cur) {
    if (wave == 0) {
      tdm_load_2d((uint32_t)(2 * cur * BM * PK),
                  Ab + (long long)m0 * lda + k0,
                  (uint32_t)((K - k0) >> 1), (uint32_t)(M - m0),
                  (uint32_t)(lda >> 1));
    } else if (wave == 1) {
      tdm_load_2d((uint32_t)(2 * (2 * BM + cur * BN) * PK),
                  Bb + (long long)n0 * ldb + k0,
                  (uint32_t)((K - k0) >> 1), (uint32_t)(N - n0),
                  (uint32_t)(ldb >> 1));
    }
  };
#else
  auto stage = [&](const uint16_t* base, int ld, int nrows, int r0, bool full,
                   int k0, uint16_t* dst) {
    if (full && (k0 + BK <= K)) {
      #pragma unroll
      for (int c = 0; c < 2; ++c) {
        int chunk = tid * 2 + c, r = chunk >> 2, kh = (chunk & 3) * 8;
        async_b128(base + (long long)(r0 + r) * ld + k0 + kh, &dst[r * PK + kh]);
      }
    } else {
      #pragma unroll
      for (int c = 0; c < 2; ++c) {
        int chunk = tid * 2 + c, r = chunk >> 2, kh = (chunk & 3) * 8;
        int gr = r0 + r, grc = gr < nrows ? gr : (nrows - 1);
        #pragma unroll 1
        for (int e = 0; e < 8; ++e) {
          int gk = k0 + kh + e, gkc = gk < K ? gk : (K - 1);
          uint16_t v = base[(long long)grc * ld + gkc];   // clamped, always valid
          dst[r * PK + kh + e] = (gr < nrows && gk < K) ? v : (uint16_t)0;
        }
      }
    }
  };
#endif

  // ---- software pipeline: stage(kt+1) while computing kt ----
#if USE_TDM
  stage_pair(0, 0);
#else
  stage(Ab, lda, M, m0, fullM, 0, sA);
  stage(Bb, ldb, N, n0, fullN, 0, sB);
#endif

  for (int kt = 0; kt < ksteps; ++kt) {
    const int cur = kt & 1;
    uint16_t* cA = sA + cur * (BM * PK);
    uint16_t* cB = sB + cur * (BN * PK);

#if USE_TDM
    wait_tdm_done();          // TENSORcnt==0 already for non-issuing waves
#else
    wait_async_done();
#endif
    __syncthreads();

    if (kt + 1 < ksteps) {
#if USE_TDM
      stage_pair((kt + 1) * BK, 1 - cur);
#else
      stage(Ab, lda, M, m0, fullM, (kt + 1) * BK, sA + (1 - cur) * (BM * PK));
      stage(Bb, ldb, N, n0, fullN, (kt + 1) * BK, sB + (1 - cur) * (BN * PK));
#endif
    }

    // A frag (16x32): lanes 0-15 K 0..7 & 16..23; lanes 16-31 K 8..15 & 24..31
    v16bf afr[2];
    #pragma unroll
    for (int tm = 0; tm < 2; ++tm) {
      int r = wm * 32 + tm * 16 + l15;
      v8u lo  = *(const v8u*)&cA[r * PK + hi * 8];
      v8u hi8 = *(const v8u*)&cA[r * PK + 16 + hi * 8];
      afr[tm] = make_frag(lo, hi8);
    }
    // B frag (32x16): col n = lane%16; lanes 0-15 K 0..7 & 16..23; lanes 16-31 K 8..15 & 24..31
    v16bf bfr[4];
    #pragma unroll
    for (int tn = 0; tn < 4; ++tn) {
      int n = wn * 64 + tn * 16 + l15;
      v8u lo  = *(const v8u*)&cB[n * PK + hi * 8];
      v8u hi8 = *(const v8u*)&cB[n * PK + 16 + hi * 8];
      bfr[tn] = make_frag(lo, hi8);
    }
    #pragma unroll
    for (int tm = 0; tm < 2; ++tm)
      #pragma unroll
      for (int tn = 0; tn < 4; ++tn)
        acc[tm][tn] = __builtin_amdgcn_wmma_f32_16x16x32_bf16(
            false, afr[tm], false, bfr[tn], (short)0, acc[tm][tn], false, false);
  }

  // ---- epilogue ----
  if (fullM && fullN) {
    #pragma unroll
    for (int tm = 0; tm < 2; ++tm)
      #pragma unroll
      for (int tn = 0; tn < 4; ++tn) {
        int col = n0 + wn * 64 + tn * 16 + l15;
        float bv = bias ? bias[col] : 0.0f;
        #pragma unroll
        for (int r = 0; r < 8; ++r) {
          int row = m0 + wm * 32 + tm * 16 + hi * 8 + r;
          float v = acc[tm][tn][r] * scale + bv;
          if (RELU) v = v > 0.0f ? v : 0.0f;
          long long idx = cbase + (long long)row * ldc + col;
          if (OUT_BF16) ((uint16_t*)Cv)[idx] = f32_to_bf16_bits(v);
          else          ((float*)Cv)[idx]    = v;
        }
      }
  } else {
    #pragma unroll
    for (int tm = 0; tm < 2; ++tm)
      #pragma unroll
      for (int tn = 0; tn < 4; ++tn) {
        int col = n0 + wn * 64 + tn * 16 + l15;
        float bv = (bias && col < N) ? bias[col] : 0.0f;
        #pragma unroll
        for (int r = 0; r < 8; ++r) {
          int row = m0 + wm * 32 + tm * 16 + hi * 8 + r;
          if (row < M && col < N) {
            float v = acc[tm][tn][r] * scale + bv;
            if (RELU) v = v > 0.0f ? v : 0.0f;
            long long idx = cbase + (long long)row * ldc + col;
            if (OUT_BF16) ((uint16_t*)Cv)[idx] = f32_to_bf16_bits(v);
            else          ((float*)Cv)[idx]    = v;
          }
        }
      }
  }
}

// ---------- embedding + positional encoding ----------
__global__ void k_embed(const float* __restrict__ x, const float* __restrict__ in_w,
                        const float* __restrict__ in_b, float* __restrict__ h,
                        uint16_t* __restrict__ hbf) {
  int idx = blockIdx.x * blockDim.x + threadIdx.x;
  if (idx >= 10752 * 512) return;
  int row = idx >> 9;
  int d   = idx & 511;
  int s   = row % 168;
  const float* xr = x + row * 10;
  float acc = in_b[d];
  #pragma unroll
  for (int k = 0; k < 10; ++k) acc += xr[k] * in_w[k * 512 + d];
  acc *= 22.627416997969522f;                       // sqrt(512)
  float di = (float)(d & ~1);
  float angle = (float)s * __expf(-9.210340371976184f * di / 512.0f);
  acc += (d & 1) ? __cosf(angle) : __sinf(angle);
  h[idx]   = acc;
  hbf[idx] = f32_to_bf16_bits(acc);
}

// ---------- masked softmax: one wave per row ----------
__global__ __launch_bounds__(256) void k_softmax(const float* __restrict__ scores,
                                                 uint16_t* __restrict__ P) {
  int gw   = blockIdx.x * 8 + (threadIdx.x >> 5);
  int lane = threadIdx.x & 31;
  int row  = gw % 168;
  long long base = (long long)(gw / 168) * (168 * 168) + (long long)row * 168;
  const float* sr = scores + base;
  uint16_t*    pr = P + base;

  float vals[6];
  float mx = -3.0e38f;
  #pragma unroll
  for (int i = 0; i < 6; ++i) {
    int c = lane + i * 32;
    float v = -1.0e30f;
    if (c < 168) { v = sr[c]; if (c > row) v += -1.0e9f; }
    vals[i] = v;
    mx = fmaxf(mx, v);
  }
  for (int off = 16; off; off >>= 1) mx = fmaxf(mx, __shfl_xor(mx, off, 32));
  float sum = 0.0f;
  #pragma unroll
  for (int i = 0; i < 6; ++i) {
    int c = lane + i * 32;
    float e = 0.0f;
    if (c < 168) e = __expf(vals[i] - mx);
    vals[i] = e; sum += e;
  }
  for (int off = 16; off; off >>= 1) sum += __shfl_xor(sum, off, 32);
  float inv = 1.0f / sum;
  #pragma unroll
  for (int i = 0; i < 6; ++i) {
    int c = lane + i * 32;
    if (c < 168) pr[c] = f32_to_bf16_bits(vals[i] * inv);
  }
}

// ---------- residual add + LayerNorm (D=512) ----------
__global__ __launch_bounds__(256) void k_add_ln(const float* __restrict__ a,
                                                const float* __restrict__ b,
                                                const float* __restrict__ g,
                                                const float* __restrict__ be,
                                                float* __restrict__ out,
                                                uint16_t* __restrict__ outbf) {
  __shared__ float red[256];
  int row = blockIdx.x;
  int tid = threadIdx.x;
  const float* ar = a + (long long)row * 512;
  const float* br = b + (long long)row * 512;
  float x0 = ar[tid] + br[tid];
  float x1 = ar[tid + 256] + br[tid + 256];
  red[tid] = x0 + x1; __syncthreads();
  for (int off = 128; off; off >>= 1) { if (tid < off) red[tid] += red[tid + off]; __syncthreads(); }
  float mean = red[0] * (1.0f / 512.0f);
  __syncthreads();
  float d0 = x0 - mean, d1 = x1 - mean;
  red[tid] = d0 * d0 + d1 * d1; __syncthreads();
  for (int off = 128; off; off >>= 1) { if (tid < off) red[tid] += red[tid + off]; __syncthreads(); }
  float inv = rsqrtf(red[0] * (1.0f / 512.0f) + 1e-9f);
  float y0 = d0 * inv * g[tid]       + be[tid];
  float y1 = d1 * inv * g[tid + 256] + be[tid + 256];
  long long o = (long long)row * 512 + tid;
  out[o] = y0; out[o + 256] = y1;
  outbf[o] = f32_to_bf16_bits(y0); outbf[o + 256] = f32_to_bf16_bits(y1);
}

// ---------- final projection [512 -> 1], keep last 48 steps ----------
__global__ __launch_bounds__(256) void k_outproj(const float* __restrict__ h,
                                                 const float* __restrict__ w,
                                                 const float* __restrict__ b,
                                                 float* __restrict__ out) {
  int gw   = blockIdx.x * 8 + (threadIdx.x >> 5);
  int lane = threadIdx.x & 31;
  if (gw >= 64 * 48) return;
  int bb = gw / 48, j = gw % 48;
  int s  = 168 - 48 + j;
  const float* hr = h + ((long long)bb * 168 + s) * 512;
  float acc = 0.0f;
  for (int k = lane; k < 512; k += 32) acc += hr[k] * w[k];
  for (int off = 16; off; off >>= 1) acc += __shfl_xor(acc, off, 32);
  if (lane == 0) out[gw] = acc + b[0];
}

// ---------- launch ----------
extern "C" void kernel_launch(void* const* d_in, const int* in_sizes, int n_in,
                              void* d_out, int out_size, void* d_ws, size_t ws_size,
                              hipStream_t stream) {
  (void)in_sizes; (void)n_in; (void)out_size; (void)ws_size;
  const float* x     = (const float*)d_in[0];
  const float* in_w  = (const float*)d_in[2];
  const float* in_b  = (const float*)d_in[3];
  const float* wq_w  = (const float*)d_in[4];
  const float* wq_b  = (const float*)d_in[5];
  const float* wk_w  = (const float*)d_in[6];
  const float* wk_b  = (const float*)d_in[7];
  const float* wv_w  = (const float*)d_in[8];
  const float* wv_b  = (const float*)d_in[9];
  const float* dn_w  = (const float*)d_in[10];
  const float* dn_b  = (const float*)d_in[11];
  const float* mh_w  = (const float*)d_in[12];
  const float* mh_b  = (const float*)d_in[13];
  const float* mo_w  = (const float*)d_in[14];
  const float* mo_b  = (const float*)d_in[15];
  const float* ln1_g = (const float*)d_in[16];
  const float* ln1_b = (const float*)d_in[17];
  const float* ln3_g = (const float*)d_in[18];
  const float* ln3_b = (const float*)d_in[19];
  const float* out_w = (const float*)d_in[20];
  const float* out_b = (const float*)d_in[21];
  float* out = (float*)d_out;

  const int R = 64 * 168;                 // 10752 rows
  const long long WQKV = 4LL * 512 * 4096;
  const long long WMH  = 4LL * 512 * 2048;

  char* p = (char*)d_ws;
  auto alloc = [&](size_t bytes) -> char* {
    char* r = p; p += (bytes + 255) & ~(size_t)255; return r;
  };
  uint16_t* wqT   = (uint16_t*)alloc(WQKV * 2);   // [L][4096][512]
  uint16_t* wkT   = (uint16_t*)alloc(WQKV * 2);
  uint16_t* wvT   = (uint16_t*)alloc(WQKV * 2);
  uint16_t* dnT   = (uint16_t*)alloc(WQKV * 2);   // [L][512][4096]
  uint16_t* mhT   = (uint16_t*)alloc(WMH * 2);    // [L][2048][512]
  uint16_t* moT   = (uint16_t*)alloc(WMH * 2);    // [L][512][2048]
  float*    h     = (float*)   alloc((size_t)R * 512 * 4);
  uint16_t* h_bf  = (uint16_t*)alloc((size_t)R * 512 * 2);
  float*    h1    = (float*)   alloc((size_t)R * 512 * 4);
  uint16_t* h1_bf = (uint16_t*)alloc((size_t)R * 512 * 2);
  float*    tmp   = (float*)   alloc((size_t)R * 512 * 4);
  uint16_t* q_bf  = (uint16_t*)alloc((size_t)R * 4096 * 2);
  uint16_t* kk_bf = (uint16_t*)alloc((size_t)R * 4096 * 2);
  uint16_t* v_bf  = (uint16_t*)alloc((size_t)R * 4096 * 2);
  uint16_t* vT    = (uint16_t*)alloc((size_t)512 * 512 * 168 * 2);  // [z][512][168]
  float*    sc    = (float*)   alloc((size_t)512 * 168 * 168 * 4);
  uint16_t* p_bf  = (uint16_t*)alloc((size_t)512 * 168 * 168 * 2);
  uint16_t* att_bf= (uint16_t*)alloc((size_t)R * 4096 * 2);
  uint16_t* hid_bf= (uint16_t*)alloc((size_t)R * 2048 * 2);

  // weight transposes (fp32 -> bf16, [K][N] -> [N][K]), once per call
  k_cvt_transpose<<<dim3(128, 16, 4), 256, 0, stream>>>(wq_w, wqT, 512, 4096, 4096, 512);
  k_cvt_transpose<<<dim3(128, 16, 4), 256, 0, stream>>>(wk_w, wkT, 512, 4096, 4096, 512);
  k_cvt_transpose<<<dim3(128, 16, 4), 256, 0, stream>>>(wv_w, wvT, 512, 4096, 4096, 512);
  k_cvt_transpose<<<dim3(16, 128, 4), 256, 0, stream>>>(dn_w, dnT, 4096, 512, 512, 4096);
  k_cvt_transpose<<<dim3(64, 16, 4),  256, 0, stream>>>(mh_w, mhT, 512, 2048, 2048, 512);
  k_cvt_transpose<<<dim3(16, 64, 4),  256, 0, stream>>>(mo_w, moT, 2048, 512, 512, 2048);

  k_embed<<<(R * 512) / 256, 256, 0, stream>>>(x, in_w, in_b, h, h_bf);

  const float inv_sqrt_d = 0.04419417382415922f;  // 1/sqrt(512)

  for (int i = 0; i < 4; ++i) {
    const long long wOff  = (long long)i * 512 * 4096;
    const long long mhOff = (long long)i * 512 * 2048;

    // Q, K, V projections: [10752 x 512] x [4096 x 512]^T -> bf16
    {
      dim3 grid(4096 / 128, R / 128, 1);
      k_gemm<true, false><<<grid, 256, 0, stream>>>(
          h_bf, wqT + wOff, wq_b + (long long)i * 4096, q_bf,
          R, 4096, 512, 512, 512, 4096, 0, 0, 0, 0, 0, 0, 1.0f);
      k_gemm<true, false><<<grid, 256, 0, stream>>>(
          h_bf, wkT + wOff, wk_b + (long long)i * 4096, kk_bf,
          R, 4096, 512, 512, 512, 4096, 0, 0, 0, 0, 0, 0, 1.0f);
      k_gemm<true, false><<<grid, 256, 0, stream>>>(
          h_bf, wvT + wOff, wv_b + (long long)i * 4096, v_bf,
          R, 4096, 512, 512, 512, 4096, 0, 0, 0, 0, 0, 0, 1.0f);
    }
    k_transpose_v<<<dim3(16, 6, 512), 256, 0, stream>>>(v_bf, vT);
    // scores = Q K^T / sqrt(512) : batched over z = b*8+h (512)
    {
      dim3 grid(2, 2, 512);
      k_gemm<false, false><<<grid, 256, 0, stream>>>(
          q_bf, kk_bf, nullptr, sc,
          168, 168, 512, 4096, 4096, 168,
          168LL * 4096, 512LL, 168LL * 4096, 512LL,
          8LL * 168 * 168, 168LL * 168, inv_sqrt_d);
    }
    k_softmax<<<(512 * 168) / 8, 256, 0, stream>>>(sc, p_bf);
    // att = P V : B = vT [z][512][168]; merge heads via ldc=4096 + head offset
    {
      dim3 grid(4, 2, 512);
      k_gemm<true, false><<<grid, 256, 0, stream>>>(
          p_bf, vT, nullptr, att_bf,
          168, 512, 168, 168, 168, 4096,
          8LL * 168 * 168, 168LL * 168, 8LL * 512 * 168, 512LL * 168,
          168LL * 4096, 512LL, 1.0f);
    }
    // dense: [10752 x 4096] x [512 x 4096]^T -> f32 tmp
    {
      dim3 grid(4, R / 128, 1);
      k_gemm<false, false><<<grid, 256, 0, stream>>>(
          att_bf, dnT + wOff, dn_b + (long long)i * 512, tmp,
          R, 512, 4096, 4096, 4096, 512, 0, 0, 0, 0, 0, 0, 1.0f);
    }
    k_add_ln<<<R, 256, 0, stream>>>(tmp, h, ln1_g + (long long)i * 512,
                                    ln1_b + (long long)i * 512, h1, h1_bf);
    // MLP up: relu(h1 @ mh + b) -> bf16
    {
      dim3 grid(16, R / 128, 1);
      k_gemm<true, true><<<grid, 256, 0, stream>>>(
          h1_bf, mhT + mhOff, mh_b + (long long)i * 2048, hid_bf,
          R, 2048, 512, 512, 512, 2048, 0, 0, 0, 0, 0, 0, 1.0f);
    }
    // MLP down -> f32 tmp
    {
      dim3 grid(4, R / 128, 1);
      k_gemm<false, false><<<grid, 256, 0, stream>>>(
          hid_bf, moT + mhOff, mo_b + (long long)i * 512, tmp,
          R, 512, 2048, 2048, 2048, 512, 0, 0, 0, 0, 0, 0, 1.0f);
    }
    k_add_ln<<<R, 256, 0, stream>>>(tmp, h1, ln3_g + (long long)i * 512,
                                    ln3_b + (long long)i * 512, h, h_bf);
  }

  k_outproj<<<(64 * 48 + 7) / 8, 256, 0, stream>>>(h, out_w, out_b, out);
}
